// KiyEngineV3_49641232007624
// MI455X (gfx1250) — matrix operand order repaired and checked by reading, not compiled, not run on hardware
//
#include <hip/hip_runtime.h>

// ---------------------------------------------------------------------------
// Problem constants (from reference)
// ---------------------------------------------------------------------------
constexpr int BB  = 4;      // batch
constexpr int LL  = 256;    // seq len
constexpr int VV  = 4096;   // vocab
constexpr int DM  = 1024;   // d_model
constexpr int DI  = 2048;   // d_inner
constexpr int DCC = 4;      // conv taps
constexpr int EE  = 8;      // experts
constexpr int NLL = 2;      // layers
constexpr int TT  = BB * LL; // 1024 tokens

typedef unsigned short u16;
typedef __bf16 bf16_t;
typedef bf16_t v16bf __attribute__((ext_vector_type(16)));
typedef float  v8f   __attribute__((ext_vector_type(8)));

__device__ __forceinline__ float bf2f(u16 u) {
    unsigned x = ((unsigned)u) << 16;
    float f;
    __builtin_memcpy(&f, &x, 4);
    return f;
}
__device__ __forceinline__ u16 f2bf(float f) {
    unsigned u;
    __builtin_memcpy(&u, &f, 4);
    unsigned lsb = (u >> 16) & 1u;
    u += 0x7fffu + lsb;            // round-to-nearest-even
    return (u16)(u >> 16);
}
__device__ __forceinline__ float silu(float x) {
    return x / (1.f + __expf(-x));
}

// ---------------------------------------------------------------------------
// Embedding gather: x[t][:] = embedding[ids[t]][:]
// ---------------------------------------------------------------------------
__global__ void k_embed(const int* __restrict__ ids, const float* __restrict__ emb,
                        float* __restrict__ x) {
    int t = blockIdx.x;
    int row = ids[t];
    const float4* s = reinterpret_cast<const float4*>(emb + (size_t)row * DM);
    float4* d = reinterpret_cast<float4*>(x + (size_t)t * DM);
    d[threadIdx.x] = s[threadIdx.x];   // 256 threads * 4 floats = 1024
}

// ---------------------------------------------------------------------------
// RMSNorm: one block (256 thr) per output row.
// src row = blockIdx.x * ROWMUL + ROWADD; MODE 0 -> bf16 out, 1 -> f32 out
// ---------------------------------------------------------------------------
template<int MODE, int ROWMUL, int ROWADD>
__global__ void k_rmsnorm(const float* __restrict__ x, const float* __restrict__ w,
                          u16* __restrict__ outB, float* __restrict__ outF) {
    __shared__ float red[256];
    int r = blockIdx.x;
    const float* xr = x + (size_t)(r * ROWMUL + ROWADD) * DM;
    float vals[4];
    float s = 0.f;
#pragma unroll
    for (int k = 0; k < 4; ++k) {
        vals[k] = xr[threadIdx.x + k * 256];
        s += vals[k] * vals[k];
    }
    red[threadIdx.x] = s;
    __syncthreads();
    for (int off = 128; off > 0; off >>= 1) {
        if (threadIdx.x < off) red[threadIdx.x] += red[threadIdx.x + off];
        __syncthreads();
    }
    float nrm = sqrtf(red[0] / (float)DM);
    float sc  = 1.f / (nrm + 1e-5f);
#pragma unroll
    for (int k = 0; k < 4; ++k) {
        int d = threadIdx.x + k * 256;
        float o = vals[k] * sc * w[d];
        if constexpr (MODE == 0) outB[(size_t)r * DM + d] = f2bf(o);
        else                     outF[(size_t)r * DM + d] = o;
    }
}

// ---------------------------------------------------------------------------
// fp32 -> bf16 conversion (grid-stride)
// ---------------------------------------------------------------------------
__global__ void k_cvt(const float* __restrict__ s, u16* __restrict__ d, int n) {
    int i = blockIdx.x * blockDim.x + threadIdx.x;
    int stride = gridDim.x * blockDim.x;
    for (; i < n; i += stride) d[i] = f2bf(s[i]);
}

// ---------------------------------------------------------------------------
// Router: logits -> softmax -> top-2 -> normalized combine weights + counts
// One thread per token (E=8, DM=1024 -> trivial work).
// ---------------------------------------------------------------------------
__global__ void k_router(const u16* __restrict__ xn, const float* __restrict__ rw,
                         const float* __restrict__ rb, float* __restrict__ comb,
                         float* __restrict__ counts) {
    int t = blockIdx.x * blockDim.x + threadIdx.x;
    if (t >= TT) return;
    float acc[EE];
#pragma unroll
    for (int e = 0; e < EE; ++e) acc[e] = rb[e];
    const u16* xr = xn + (size_t)t * DM;
    for (int d = 0; d < DM; ++d) {
        float xv = bf2f(xr[d]);
#pragma unroll
        for (int e = 0; e < EE; ++e) acc[e] += xv * rw[(size_t)e * DM + d];
    }
    float m = acc[0];
#pragma unroll
    for (int e = 1; e < EE; ++e) m = fmaxf(m, acc[e]);
    float p[EE];
#pragma unroll
    for (int e = 0; e < EE; ++e) p[e] = __expf(acc[e] - m);
    int i1 = 0;
#pragma unroll
    for (int e = 1; e < EE; ++e) if (p[e] > p[i1]) i1 = e;
    int i2 = (i1 == 0) ? 1 : 0;
#pragma unroll
    for (int e = 0; e < EE; ++e) if (e != i1 && p[e] > p[i2]) i2 = e;
    float inv = 1.f / (p[i1] + p[i2]);
#pragma unroll
    for (int e = 0; e < EE; ++e) comb[(size_t)t * EE + e] = 0.f;
    comb[(size_t)t * EE + i1] = p[i1] * inv;
    comb[(size_t)t * EE + i2] = p[i2] * inv;
    atomicAdd(&counts[i1], 1.f);
    atomicAdd(&counts[i2], 1.f);
}

// ---------------------------------------------------------------------------
// WMMA bf16 GEMM:  C[M,N] = sum_e A[:, e*ASTRIDE + 0..KIN) * B_e[N,KIN]^T
//   Template params make every loop bound / branch compile-time so the WMMA
//   region stays fully convergent (EXEC all-1s, no exec-mask churn).
//   Block tile 128x128, 8 wave32 waves (2x4), wave 64x32 = 4x2 wmma tiles.
//   Double-buffered LDS: stage tile k+1 while computing tile k.
//   MODE 0: Cf[o] = acc + Cadd[o]   (f32 out, fused residual)
//   MODE 1: Cb[o] = bf16(acc)       (bf16 out)
// ---------------------------------------------------------------------------
#define LDSK 40   // padded K stride (ushorts): 80B rows, 16B-aligned chunks

template<int NUME, int KIN, int ASTRIDE, int MODE>
__global__ __launch_bounds__(256) void k_gemm_wmma(
    const u16* __restrict__ A, int lda,
    const u16* __restrict__ Bw, int ldb, size_t bStrideE,
    float* __restrict__ Cf, u16* __restrict__ Cb,
    const float* __restrict__ Cadd, int ldc)
{
    __shared__ u16 sA[2][128 * LDSK];
    __shared__ u16 sB[2][128 * LDSK];

    const int tid  = threadIdx.x;
    const int lane = tid & 31;
    const int wid  = tid >> 5;
    const int wm   = wid >> 2;     // 0..1 (64-row slab)
    const int wn   = wid & 3;      // 0..3 (32-col slab)
    const int ln   = lane & 15;
    const int hi   = lane >> 4;    // lane half selects K sub-range
    const int n0   = blockIdx.x * 128;
    const int m0   = blockIdx.y * 128;

    // branch-free staging: each thread moves one 32B chunk of A and of B
    const int srow = tid >> 1;           // 0..127
    const int skc  = (tid & 1) * 16;     // 0 or 16 (bf16 elems)

    auto stage = [&](int e, int kt, int b) {
        const u16* ga = A + (size_t)(m0 + srow) * lda + (size_t)e * ASTRIDE + kt + skc;
        const u16* gb = Bw + (size_t)e * bStrideE
                           + (size_t)(n0 + srow) * ldb + kt + skc;
        uint4 a0 = *reinterpret_cast<const uint4*>(ga);
        uint4 a1 = *reinterpret_cast<const uint4*>(ga + 8);
        uint4 b0 = *reinterpret_cast<const uint4*>(gb);
        uint4 b1 = *reinterpret_cast<const uint4*>(gb + 8);
        u16* da = &sA[b][srow * LDSK + skc];
        u16* db = &sB[b][srow * LDSK + skc];
        *reinterpret_cast<uint4*>(da)     = a0;
        *reinterpret_cast<uint4*>(da + 8) = a1;
        *reinterpret_cast<uint4*>(db)     = b0;
        *reinterpret_cast<uint4*>(db + 8) = b1;
    };

    v8f acc[4][2];
#pragma unroll
    for (int i = 0; i < 4; ++i)
#pragma unroll
        for (int j = 0; j < 2; ++j)
#pragma unroll
            for (int r = 0; r < 8; ++r) acc[i][j][r] = 0.f;

    constexpr int KSTEPS = KIN / 32;
    constexpr int NITER  = NUME * KSTEPS;

    stage(0, 0, 0);
    int buf = 0;
    for (int it = 0; it < NITER; ++it) {
        __syncthreads();                       // staged data for `buf` visible
        if (it + 1 < NITER) {                  // uniform scalar branch
            int nx = it + 1;
            stage(nx / KSTEPS, (nx % KSTEPS) * 32, buf ^ 1);
        }

        // A fragments: 16x32 bf16; lanes 0-15 rows; K pairs interleaved by
        // lane-half (K 0-7,16-23 vs 8-15,24-31) per the CDNA5 VGPR layout.
        v16bf afr[4];
#pragma unroll
        for (int mt = 0; mt < 4; ++mt) {
            const u16* base = &sA[buf][(wm * 64 + mt * 16 + ln) * LDSK];
            union { v16bf v; uint4 q[2]; } u;
            u.q[0] = *reinterpret_cast<const uint4*>(base + hi * 8);
            u.q[1] = *reinterpret_cast<const uint4*>(base + hi * 8 + 16);
            afr[mt] = u.v;
        }
        // B fragments: 32x16 bf16; lane holds a column; K split by lane-half (0-7,16-23 vs 8-15,24-31)
        v16bf bfr[2];
#pragma unroll
        for (int nt = 0; nt < 2; ++nt) {
            const u16* base = &sB[buf][(wn * 32 + nt * 16 + ln) * LDSK];
            union { v16bf v; uint4 q[2]; } u;
            u.q[0] = *reinterpret_cast<const uint4*>(base + hi * 8);
            u.q[1] = *reinterpret_cast<const uint4*>(base + hi * 8 + 16);
            bfr[nt] = u.v;
        }
#pragma unroll
        for (int mt = 0; mt < 4; ++mt)
#pragma unroll
            for (int nt = 0; nt < 2; ++nt)
                acc[mt][nt] = __builtin_amdgcn_wmma_f32_16x16x32_bf16(
                    false, afr[mt], false, bfr[nt],
                    (short)0, acc[mt][nt], false, false);
        buf ^= 1;
    }

    // Epilogue: D layout -> lane gives column, vgpr index r + lane-half give row
#pragma unroll
    for (int mt = 0; mt < 4; ++mt) {
#pragma unroll
        for (int nt = 0; nt < 2; ++nt) {
            int col = n0 + wn * 32 + nt * 16 + ln;
#pragma unroll
            for (int r = 0; r < 8; ++r) {
                int row = m0 + wm * 64 + mt * 16 + r + hi * 8;
                size_t o = (size_t)row * ldc + col;
                if constexpr (MODE == 0) Cf[o] = acc[mt][nt][r] + Cadd[o];
                else                     Cb[o] = f2bf(acc[mt][nt][r]);
            }
        }
    }
}

// ---------------------------------------------------------------------------
// Fused activation: y[t,e,h] = comb[t,e] * silu(xi*cw_last+cb) * D * silu(z)
// ---------------------------------------------------------------------------
__global__ void k_act(const u16* __restrict__ xz, const float* __restrict__ cw,
                      const float* __restrict__ cb, const float* __restrict__ Dp,
                      const float* __restrict__ comb, u16* __restrict__ y) {
    size_t idx = (size_t)blockIdx.x * 256 + threadIdx.x;  // T*E*DI total
    int h = (int)(idx % DI);
    size_t r = idx / DI;
    int e = (int)(r % EE);
    int t = (int)(r / EE);
    size_t base = (size_t)t * (EE * 2 * DI) + (size_t)e * 2 * DI;
    float xi = bf2f(xz[base + h]);
    float z  = bf2f(xz[base + DI + h]);
    float cwl = cw[((size_t)e * DI + h) * DCC + (DCC - 1)];
    float xc = xi * cwl + cb[(size_t)e * DI + h];
    float yy = silu(xc) * Dp[(size_t)e * DI + h] * silu(z) *
               comb[(size_t)t * EE + e];
    y[(size_t)t * (EE * DI) + (size_t)e * DI + h] = f2bf(yy);
}

// ---------------------------------------------------------------------------
// Policy head: out[b,v] = dot(last[b], policy_w[v]); one thread per output
// ---------------------------------------------------------------------------
__global__ void k_policy(const float* __restrict__ last, const float* __restrict__ pw,
                         float* __restrict__ out) {
    int idx = blockIdx.x * 256 + threadIdx.x;   // 16384
    int b = idx >> 12;
    int v = idx & 4095;
    const float4* x = reinterpret_cast<const float4*>(last + (size_t)b * DM);
    const float4* w = reinterpret_cast<const float4*>(pw + (size_t)v * DM);
    float s = 0.f;
    for (int i = 0; i < DM / 4; ++i) {
        float4 a = x[i], c = w[i];
        s += a.x * c.x + a.y * c.y + a.z * c.z + a.w * c.w;
    }
    out[idx] = s;
}

// ---------------------------------------------------------------------------
// Value head: tanh(relu(last@W1^T+b1)@W2^T+b2); one block of 128 per batch row
// ---------------------------------------------------------------------------
__global__ void k_value(const float* __restrict__ last, const float* __restrict__ w1,
                        const float* __restrict__ b1, const float* __restrict__ w2,
                        const float* __restrict__ b2, float* __restrict__ out) {
    __shared__ float h[128];
    int b = blockIdx.x, j = threadIdx.x;
    const float* x = last + (size_t)b * DM;
    const float* wr = w1 + (size_t)j * DM;
    float s = b1[j];
    for (int d = 0; d < DM; ++d) s += x[d] * wr[d];
    h[j] = fmaxf(s, 0.f);
    __syncthreads();
    if (j == 0) {
        float v = b2[0];
        for (int k = 0; k < 128; ++k) v += h[k] * w2[k];
        out[BB * VV + b] = tanhf(v);
    }
}

// ---------------------------------------------------------------------------
// Aux loss finalize: mean over layers of sum_e (count_e/T)^2
// ---------------------------------------------------------------------------
__global__ void k_aux(const float* __restrict__ counts, float* __restrict__ out) {
    float a = 0.f;
    for (int l = 0; l < NLL; ++l) {
        float s = 0.f;
        for (int e = 0; e < EE; ++e) {
            float ld = counts[l * EE + e] / (float)TT;
            s += ld * ld;
        }
        a += s;
    }
    out[BB * VV + BB] = a / (float)NLL;
}

// ---------------------------------------------------------------------------
// Host orchestration
// ---------------------------------------------------------------------------
extern "C" void kernel_launch(void* const* d_in, const int* in_sizes, int n_in,
                              void* d_out, int out_size, void* d_ws, size_t ws_size,
                              hipStream_t stream) {
    (void)in_sizes; (void)n_in; (void)out_size; (void)ws_size;
    const int*   ids       = (const int*)d_in[0];
    const float* emb       = (const float*)d_in[1];
    const float* norm_w    = (const float*)d_in[2];
    const float* router_w  = (const float*)d_in[3];
    const float* router_b  = (const float*)d_in[4];
    const float* in_proj_w = (const float*)d_in[5];
    const float* conv_w    = (const float*)d_in[6];
    const float* conv_b    = (const float*)d_in[7];
    const float* D_param   = (const float*)d_in[8];
    const float* out_proj_w= (const float*)d_in[9];
    const float* policy_w  = (const float*)d_in[10];
    const float* value_w1  = (const float*)d_in[11];
    const float* value_b1  = (const float*)d_in[12];
    const float* value_w2  = (const float*)d_in[13];
    const float* value_b2  = (const float*)d_in[14];
    float* out = (float*)d_out;

    char* p = (char*)d_ws;
    auto carve = [&](size_t bytes) {
        char* r = p;
        p += (bytes + 255) & ~(size_t)255;
        return r;
    };
    float* x      = (float*)carve((size_t)TT * DM * 4);
    u16*   xn     = (u16*)  carve((size_t)TT * DM * 2);
    u16*   wiB    = (u16*)  carve((size_t)EE * 2 * DI * DM * 2);
    u16*   woB    = (u16*)  carve((size_t)EE * DM * DI * 2);
    u16*   xz     = (u16*)  carve((size_t)TT * EE * 2 * DI * 2);
    u16*   y      = (u16*)  carve((size_t)TT * EE * DI * 2);
    float* comb   = (float*)carve((size_t)TT * EE * 4);
    float* counts = (float*)carve((size_t)NLL * EE * 4);
    float* last   = (float*)carve((size_t)BB * DM * 4);

    hipMemsetAsync(counts, 0, NLL * EE * sizeof(float), stream);
    k_embed<<<TT, 256, 0, stream>>>(ids, emb, x);

    for (int l = 0; l < NLL; ++l) {
        k_rmsnorm<0, 1, 0><<<TT, 256, 0, stream>>>(x, norm_w, xn, nullptr);
        k_cvt<<<4096, 256, 0, stream>>>(in_proj_w + (size_t)l * EE * 2 * DI * DM,
                                        wiB, EE * 2 * DI * DM);
        k_cvt<<<4096, 256, 0, stream>>>(out_proj_w + (size_t)l * EE * DM * DI,
                                        woB, EE * DM * DI);
        k_router<<<TT / 256, 256, 0, stream>>>(xn, router_w + (size_t)l * EE * DM,
                                               router_b + l * EE, comb,
                                               counts + l * EE);
        // in_proj: xz(T, E*2DI) = xn(T,DM) @ WiBig(E*2DI, DM)^T  -> bf16
        dim3 g1((EE * 2 * DI) / 128, TT / 128);
        k_gemm_wmma<1, DM, 0, 1><<<g1, 256, 0, stream>>>(
            xn, DM, wiB, DM, (size_t)0, nullptr, xz, nullptr, EE * 2 * DI);
        // fused gate / conv-tap / D / comb scaling -> y bf16
        k_act<<<(unsigned)(((size_t)TT * EE * DI) / 256), 256, 0, stream>>>(
            xz, conv_w + (size_t)l * EE * DI * DCC, conv_b + (size_t)l * EE * DI,
            D_param + (size_t)l * EE * DI, comb, y);
        // out_proj: x(T,DM) = sum_e y(T, e*DI..) @ Wo_e(DM,DI)^T + x  (residual)
        dim3 g2(DM / 128, TT / 128);
        k_gemm_wmma<EE, DI, DI, 0><<<g2, 256, 0, stream>>>(
            y, EE * DI, woB, DI, (size_t)DM * DI, x, nullptr, x, DM);
    }

    // final rmsnorm only on last token per batch row
    k_rmsnorm<1, LL, LL - 1><<<BB, 256, 0, stream>>>(x, norm_w, nullptr, last);
    k_policy<<<(BB * VV) / 256, 256, 0, stream>>>(last, policy_w, out);
    k_value<<<BB, 128, 0, stream>>>(last, value_w1, value_b1, value_w2, value_b2, out);
    k_aux<<<1, 1, 0, stream>>>(counts, out);
}
